// SpikeRNN_19447611916936
// MI455X (gfx1250) — matrix operand order repaired and checked by reading, not compiled, not run on hardware
//
#include <hip/hip_runtime.h>

typedef __attribute__((ext_vector_type(16))) _Float16 v16h;
typedef __attribute__((ext_vector_type(8)))  _Float16 v8h;
typedef __attribute__((ext_vector_type(8)))  float    v8f;

#define NB_IN    256
#define NB_REC   1024
#define NB_OUT   64
#define BATCH    128
#define NB_STEPS 300
#define TSTEPS   299
#define NCAT     (NB_REC + NB_OUT)   // 1088 output columns (rec | out)
#define KCAT     (NB_REC + NB_IN)    // 1280 contraction dim (spikes | inputs)
#define ALPHA_C  0.818730753078f     // exp(-0.001/0.005)
#define BETA_C   0.904837418036f     // exp(-0.001/0.010)

// ---- d_out regions (float offsets) -------------------------------------
#define OUTREC_OFF 0                         // [128,300,64]
#define SPK_OFF    2457600                   // [128,300,1024]
#define WRECD_OFF  41779200                  // [1024,1024]
#define WOUTD_OFF  42827776                  // [1024,64]
#define WFF_OFF    42893312                  // [256,1024]

// ---- workspace byte offsets (256B aligned) -----------------------------
#define WS_BT_HI   0                         // f16 [1088][1280] W^T hi
#define WS_BT_LO   2785280                   // f16 [1088][1280] W^T lo
#define WS_SYN     5570560                   // f32 [128][1024]
#define WS_MEM     6094848                   // f32 [128][1024]
#define WS_SPKH    6619136                   // f16 [128][1024]

// ========================= setup kernels ================================

// w_rec_d = d_sign[i]*|w_rec[i,j]|; also fill B^T[j][i] hi/lo halves
__global__ void k_setup_wrec(const float* __restrict__ w_rec,
                             const float* __restrict__ d_sign,
                             float* __restrict__ wrecd,
                             _Float16* __restrict__ bt_hi,
                             _Float16* __restrict__ bt_lo) {
  int idx = blockIdx.x * blockDim.x + threadIdx.x;   // 1024*1024
  int i = idx >> 10, j = idx & 1023;
  float w = d_sign[i] * fabsf(w_rec[idx]);
  wrecd[idx] = w;
  _Float16 h = (_Float16)w;
  bt_hi[(long)j * KCAT + i] = h;
  bt_lo[(long)j * KCAT + i] = (_Float16)(w - (float)h);
}

// w_out_d = d_sign[i]*|w_out[i,o]|; fill B^T rows 1024..1087, k < 1024
__global__ void k_setup_wout(const float* __restrict__ w_out,
                             const float* __restrict__ d_sign,
                             float* __restrict__ woutd,
                             _Float16* __restrict__ bt_hi,
                             _Float16* __restrict__ bt_lo) {
  int idx = blockIdx.x * blockDim.x + threadIdx.x;   // 1024*64
  int i = idx >> 6, o = idx & 63;
  float w = d_sign[i] * fabsf(w_out[idx]);
  woutd[idx] = w;
  _Float16 h = (_Float16)w;
  bt_hi[(long)(NB_REC + o) * KCAT + i] = h;
  bt_lo[(long)(NB_REC + o) * KCAT + i] = (_Float16)(w - (float)h);
}

// copy w_ff to output; fill B^T[j][1024+c] (input segment of K)
__global__ void k_setup_wff(const float* __restrict__ w_ff,
                            float* __restrict__ wffo,
                            _Float16* __restrict__ bt_hi,
                            _Float16* __restrict__ bt_lo) {
  int idx = blockIdx.x * blockDim.x + threadIdx.x;   // 256*1024
  int c = idx >> 10, j = idx & 1023;
  float w = w_ff[idx];
  wffo[idx] = w;
  _Float16 h = (_Float16)w;
  bt_hi[(long)j * KCAT + NB_REC + c] = h;
  bt_lo[(long)j * KCAT + NB_REC + c] = (_Float16)(w - (float)h);
}

// zero state, t=0 output slices, and the (out-cols x input-K) zero pad of B^T
__global__ void k_setup_zero(float* __restrict__ syn, float* __restrict__ mem,
                             float* __restrict__ spk, float* __restrict__ outrec,
                             _Float16* __restrict__ bt_hi, _Float16* __restrict__ bt_lo) {
  int idx = blockIdx.x * blockDim.x + threadIdx.x;   // 131072
  syn[idx] = 0.0f;
  mem[idx] = 0.0f;
  int b = idx >> 10, r = idx & 1023;
  spk[(long)b * NB_STEPS * NB_REC + r] = 0.0f;       // spk_rec[:,0,:]
  if (idx < BATCH * NB_OUT) {
    int b2 = idx >> 6, o = idx & 63;
    outrec[(long)b2 * NB_STEPS * NB_OUT + o] = 0.0f; // out_rec[:,0,:]
  }
  if (idx < NB_OUT * NB_IN) {                        // pad: out cols have no w_ff
    int o = idx >> 8, k = idx & 255;
    bt_hi[(long)(NB_REC + o) * KCAT + NB_REC + k] = (_Float16)0.0f;
    bt_lo[(long)(NB_REC + o) * KCAT + NB_REC + k] = (_Float16)0.0f;
  }
}

// ========================= per-step kernels =============================

// elementwise: reset, spike, membrane update (uses OLD syn), record spikes
__global__ void k_spike(const float* __restrict__ syn, float* __restrict__ mem,
                        float* __restrict__ spk, _Float16* __restrict__ spkh, int t) {
  int idx = blockIdx.x * blockDim.x + threadIdx.x;   // 131072
  int b = idx >> 10, r = idx & 1023;
  const long base = (long)b * NB_STEPS * NB_REC;
  float rst1 = spk[base + (long)t * NB_REC + r];                      // out_{t-1}
  float rst2 = spk[base + (long)(t > 0 ? t - 1 : 0) * NB_REC + r];   // out_{t-2}
  float mv = mem[idx];
  if (rst1 > 0.0f || rst2 > 0.0f) mv = 0.0f;
  float o = (mv - 1.0f) > 0.0f ? 1.0f : 0.0f;
  mem[idx] = BETA_C * mv + syn[idx] - o;
  spk[base + (long)(t + 1) * NB_REC + r] = o;
  spkh[idx] = (_Float16)o;
}

// fused GEMM: [spk | inputs_t] (128x1280) @ Bcat^T (1280x1088)
// epilogue: syn = alpha*syn + acc + noise (cols<1024), out_rec[t+1] (cols>=1024)
__global__ void __launch_bounds__(128)
k_step_gemm(const _Float16* __restrict__ spkh,
            const _Float16* __restrict__ bt_hi,
            const _Float16* __restrict__ bt_lo,
            const float* __restrict__ inputs,
            const float* __restrict__ noise,
            float* __restrict__ syn,
            float* __restrict__ outrec,
            int t) {
  const int lane = threadIdx.x & 31;
  const int wave = threadIdx.x >> 5;
  const int tile = blockIdx.x * 4 + wave;            // 0..543
  const int m_base = (tile & 7) * 16;                // 8 M tiles (batch)
  const int n_base = (tile >> 3) * 16;               // 68 N tiles (rec|out)
  const int l15 = lane & 15, lh = lane >> 4;

  const int m = m_base + l15;                        // A row owned by this lane
  const int n = n_base + l15;                        // B column owned by this lane

  const _Float16* arow   = spkh + (long)m * NB_REC;
  const _Float16* bh_row = bt_hi + (long)n * KCAT;
  const _Float16* bl_row = bt_lo + (long)n * KCAT;
  const float*    in_row = inputs + ((long)m * NB_STEPS + t) * NB_IN;

  v8f acc = {};

  // --- spike segment of K (0..1023): A is exactly representable in f16 ---
  for (int kb = 0; kb < NB_REC; kb += 32) {
    v8h a0 = *(const v8h*)(arow + kb + lh * 16);
    v8h a1 = *(const v8h*)(arow + kb + lh * 16 + 8);
    v16h a = __builtin_shufflevector(a0, a1, 0,1,2,3,4,5,6,7,8,9,10,11,12,13,14,15);
    v16h bh = *(const v16h*)(bh_row + kb + lh * 16);
    v16h bl = *(const v16h*)(bl_row + kb + lh * 16);
    acc = __builtin_amdgcn_wmma_f32_16x16x32_f16(false, a, false, bh, (short)0, acc, false, false);
    acc = __builtin_amdgcn_wmma_f32_16x16x32_f16(false, a, false, bl, (short)0, acc, false, false);
  }

  // --- input segment of K (1024..1279): split inputs hi/lo on the fly ---
  for (int kc = 0; kc < NB_IN; kc += 32) {
    v8f f0 = *(const v8f*)(in_row + kc + lh * 16);
    v8f f1 = *(const v8f*)(in_row + kc + lh * 16 + 8);
    v16h ah, al;
#pragma unroll
    for (int i = 0; i < 8; ++i) {
      float x = f0[i];
      _Float16 h = (_Float16)x;
      ah[i] = h; al[i] = (_Float16)(x - (float)h);
      float y = f1[i];
      _Float16 h2 = (_Float16)y;
      ah[8 + i] = h2; al[8 + i] = (_Float16)(y - (float)h2);
    }
    const int kb = NB_REC + kc;
    v16h bh = *(const v16h*)(bh_row + kb + lh * 16);
    v16h bl = *(const v16h*)(bl_row + kb + lh * 16);
    acc = __builtin_amdgcn_wmma_f32_16x16x32_f16(false, ah, false, bh, (short)0, acc, false, false);
    acc = __builtin_amdgcn_wmma_f32_16x16x32_f16(false, ah, false, bl, (short)0, acc, false, false);
    acc = __builtin_amdgcn_wmma_f32_16x16x32_f16(false, al, false, bh, (short)0, acc, false, false);
  }

  // --- epilogue (C/D layout: lanes0-15 M=r, lanes16-31 M=r+8; N=lane&15) ---
  const int n_g = n_base + l15;
  const int m_off = lh * 8;
  if (n_g < NB_REC) {
#pragma unroll
    for (int r = 0; r < 8; ++r) {
      int m_g = m_base + m_off + r;
      long si = (long)m_g * NB_REC + n_g;
      syn[si] = ALPHA_C * syn[si] + acc[r] +
                noise[((long)m_g * NB_STEPS + t) * NB_REC + n_g];
    }
  } else {
#pragma unroll
    for (int r = 0; r < 8; ++r) {
      int m_g = m_base + m_off + r;
      outrec[((long)m_g * NB_STEPS + (t + 1)) * NB_OUT + (n_g - NB_REC)] = acc[r];
    }
  }
}

// ========================= launch =======================================

extern "C" void kernel_launch(void* const* d_in, const int* in_sizes, int n_in,
                              void* d_out, int out_size, void* d_ws, size_t ws_size,
                              hipStream_t stream) {
  const float* inputs = (const float*)d_in[0];
  const float* noise  = (const float*)d_in[1];
  const float* w_ff   = (const float*)d_in[2];
  const float* w_rec  = (const float*)d_in[3];
  const float* w_out  = (const float*)d_in[4];
  const float* d_sign = (const float*)d_in[5];

  float* out    = (float*)d_out;
  float* outrec = out + OUTREC_OFF;
  float* spk    = out + SPK_OFF;
  float* wrecd  = out + WRECD_OFF;
  float* woutd  = out + WOUTD_OFF;
  float* wffo   = out + WFF_OFF;

  char* ws = (char*)d_ws;
  _Float16* bt_hi = (_Float16*)(ws + WS_BT_HI);
  _Float16* bt_lo = (_Float16*)(ws + WS_BT_LO);
  float*    syn   = (float*)(ws + WS_SYN);
  float*    mem   = (float*)(ws + WS_MEM);
  _Float16* spkh  = (_Float16*)(ws + WS_SPKH);

  k_setup_wrec<<<4096, 256, 0, stream>>>(w_rec, d_sign, wrecd, bt_hi, bt_lo);
  k_setup_wout<<<256, 256, 0, stream>>>(w_out, d_sign, woutd, bt_hi, bt_lo);
  k_setup_wff<<<1024, 256, 0, stream>>>(w_ff, wffo, bt_hi, bt_lo);
  k_setup_zero<<<512, 256, 0, stream>>>(syn, mem, spk, outrec, bt_hi, bt_lo);

  for (int t = 0; t < TSTEPS; ++t) {
    k_spike<<<512, 256, 0, stream>>>(syn, mem, spk, spkh, t);
    k_step_gemm<<<136, 128, 0, stream>>>(spkh, bt_hi, bt_lo, inputs, noise,
                                         syn, outrec, t);
  }
}
